// MultiHeadAttention_26542897889606
// MI455X (gfx1250) — compile-verified, matrix operand order checked by reading, not run on hardware
//
#include <hip/hip_runtime.h>
#include <cstdint>
#include <cstddef>

// ---------------------------------------------------------------------------
// MHA forward for B=2, S=2048, H=16, DH=64, D=1024 (fp32 in/out, bf16 WMMA)
// Outputs (concatenated in d_out): output [B,S,D] f32, attn [B,H,S,S] f32
// All GEMMs: B operand is n-major [N][K] so LDS staging is contiguous 16B
// chunks -> async global->LDS (double buffered), consumed by v_wmma bf16.
// ---------------------------------------------------------------------------

typedef __attribute__((ext_vector_type(16))) __bf16 v16bf;
typedef __attribute__((ext_vector_type(8)))  float  v8f;

#define B_  2
#define S_  2048
#define H_  16
#define DH_ 64
#define D_  1024

__device__ __forceinline__ unsigned short f2bf(float f) {
  unsigned u = __float_as_uint(f);
  u += 0x7FFFu + ((u >> 16) & 1u);   // round-to-nearest-even
  return (unsigned short)(u >> 16);
}
__device__ __forceinline__ unsigned pack2(float a, float b) {
  return (unsigned)f2bf(a) | ((unsigned)f2bf(b) << 16);
}

// CDNA5 async global->LDS copy (ASYNCcnt-tracked), 16 bytes per lane.
// vdst = VGPR holding LDS byte offset, vaddr = 64-bit global address, GV mode.
__device__ __forceinline__ void async_g2l_b128(unsigned lds_off, const void* g) {
  asm volatile("global_load_async_to_lds_b128 %0, %1, off"
               :: "v"(lds_off), "v"((unsigned long long)(uintptr_t)g)
               : "memory");
}
template <int N>
__device__ __forceinline__ void wait_asynccnt() {
  asm volatile("s_wait_asynccnt %0" :: "i"(N) : "memory");
}

// ---------------------------------------------------------------------------
// f32 -> bf16 cast, 4 elements/thread (activations, kept row-major)
// ---------------------------------------------------------------------------
__global__ void cast_f32_bf16(const float* __restrict__ src,
                              unsigned short* __restrict__ dst, int n4) {
  int i = blockIdx.x * blockDim.x + threadIdx.x;
  if (i >= n4) return;
  float4 f = ((const float4*)src)[i];
  uint2 o;
  o.x = pack2(f.x, f.y);
  o.y = pack2(f.z, f.w);
  ((uint2*)dst)[i] = o;
}

// ---------------------------------------------------------------------------
// Weight transpose-cast: W f32 [K][N] -> bf16 [N][K] (LDS-tiled 32x32)
// ---------------------------------------------------------------------------
__global__ void cast_transpose_w(const float* __restrict__ src,
                                 unsigned short* __restrict__ dst) {
  __shared__ float t[32][33];
  const int tx = threadIdx.x & 31;
  const int ty = threadIdx.x >> 5;          // 0..7
  const int k0 = blockIdx.x * 32;           // source row block (K)
  const int n0 = blockIdx.y * 32;           // source col block (N)
#pragma unroll
  for (int j = 0; j < 4; ++j) {
    const int r = ty + j * 8;
    t[r][tx] = src[(size_t)(k0 + r) * D_ + n0 + tx];
  }
  __syncthreads();
#pragma unroll
  for (int j = 0; j < 4; ++j) {
    const int r = ty + j * 8;               // local n
    dst[(size_t)(n0 + r) * D_ + k0 + tx] = f2bf(t[tx][r]);
  }
}

// ---------------------------------------------------------------------------
// Row softmax, in place over attn [B*H*S rows, S cols]; 256 thr, 8 elem/thr
// ---------------------------------------------------------------------------
__global__ void softmax_rows(float* __restrict__ attn) {
  const int tid = threadIdx.x;
  float* p = attn + (size_t)blockIdx.x * S_;
  __shared__ float red[256];

  float v[8];
  float m = -3.4e38f;
#pragma unroll
  for (int j = 0; j < 8; ++j) {
    v[j] = p[tid + j * 256];
    m = fmaxf(m, v[j]);
  }
  red[tid] = m; __syncthreads();
  for (int s = 128; s > 0; s >>= 1) {
    if (tid < s) red[tid] = fmaxf(red[tid], red[tid + s]);
    __syncthreads();
  }
  m = red[0]; __syncthreads();

  float sum = 0.f;
#pragma unroll
  for (int j = 0; j < 8; ++j) { v[j] = __expf(v[j] - m); sum += v[j]; }
  red[tid] = sum; __syncthreads();
  for (int s = 128; s > 0; s >>= 1) {
    if (tid < s) red[tid] += red[tid + s];
    __syncthreads();
  }
  const float inv = 1.0f / red[0];
#pragma unroll
  for (int j = 0; j < 8; ++j) p[tid + j * 256] = v[j] * inv;
}

// ---------------------------------------------------------------------------
// WMMA bf16 GEMM: block tile 128(M) x 64(N), K-step 32, 8 waves (wave32).
// Wave grid 4x2, each wave computes 32x32 = 2x2 v_wmma_f32_16x16x32_bf16.
// B is ALWAYS n-major [N][K]. Async global->LDS, double-buffered K pipeline.
// MODE 0: A*Wt + bias -> bf16 head-split [B,H,S,DH]       (Q,K proj)
// MODE 1: A*Wt + bias -> bf16 head-split transposed [B,H,DH,S]  (V proj)
// MODE 2: A*Wt + bias -> f32 plain [M,D]                  (out proj)
// MODE 3: Q*K^T       -> f32 scaled+masked scores, z=bh
// MODE 4: attn(f32)*Vt-> bf16 context [B,S,D], z=bh       (A is f32)
// ---------------------------------------------------------------------------
enum { MODE_QK = 0, MODE_VT = 1, MODE_OUTPROJ = 2, MODE_SCORES = 3, MODE_CONTEXT = 4 };

template <int MODE, bool A_F32>
__global__ __launch_bounds__(256) void gemm_wmma(
    const void* __restrict__ Av, const unsigned short* __restrict__ Bmat,
    const float* __restrict__ bias, const unsigned char* __restrict__ mask,
    void* __restrict__ outv, int M, int N, int K, int lda, int ldb) {
  __shared__ unsigned short As[2][128 * 32];   // [m][k]
  __shared__ unsigned short Bs[2][64 * 32];    // [n][k]

  const int tid   = threadIdx.x;
  const int lane  = tid & 31;
  const int w     = tid >> 5;
  const int waveM = w & 3;
  const int waveN = w >> 2;
  const int m0    = blockIdx.y * 128;
  const int n0    = blockIdx.x * 64;
  const int z     = blockIdx.z;

  const unsigned short* Abf = (const unsigned short*)Av;
  const float*          Af  = (const float*)Av;
  size_t aOff = 0, bOff = 0;
  if (MODE == MODE_SCORES)  { aOff = (size_t)z * S_ * DH_; bOff = (size_t)z * S_ * DH_; }
  if (MODE == MODE_CONTEXT) { aOff = (size_t)z * S_ * S_;  bOff = (size_t)z * S_ * DH_; }

  v8f acc[2][2];
#pragma unroll
  for (int mi = 0; mi < 2; ++mi)
#pragma unroll
    for (int ni = 0; ni < 2; ++ni)
#pragma unroll
      for (int r = 0; r < 8; ++r) acc[mi][ni][r] = 0.f;

  // per-stage staging: A 128x32 bf16 (32B/thread), B 64x32 bf16 (16B/thread)
  const int arow = tid >> 1, ahalf = tid & 1;
  const int bn   = tid >> 2, bq    = tid & 3;

  auto issue = [&](int k0, int buf) {
    if (!A_F32) {
      const unsigned short* gp =
          Abf + aOff + (size_t)(m0 + arow) * lda + k0 + ahalf * 16;
      unsigned lo = (unsigned)(uintptr_t)&As[buf][arow * 32 + ahalf * 16];
      async_g2l_b128(lo,      gp);
      async_g2l_b128(lo + 16, gp + 8);
    } else {
      // context: convert f32 attn -> bf16 while staging (synchronous)
      const float* gp = Af + aOff + (size_t)(m0 + arow) * lda + k0 + ahalf * 16;
      float4 f0 = ((const float4*)gp)[0];
      float4 f1 = ((const float4*)gp)[1];
      float4 f2 = ((const float4*)gp)[2];
      float4 f3 = ((const float4*)gp)[3];
      uint4 o0, o1;
      o0.x = pack2(f0.x, f0.y); o0.y = pack2(f0.z, f0.w);
      o0.z = pack2(f1.x, f1.y); o0.w = pack2(f1.z, f1.w);
      o1.x = pack2(f2.x, f2.y); o1.y = pack2(f2.z, f2.w);
      o1.z = pack2(f3.x, f3.y); o1.w = pack2(f3.z, f3.w);
      uint4* d = (uint4*)&As[buf][arow * 32 + ahalf * 16];
      d[0] = o0; d[1] = o1;
    }
    {
      const unsigned short* gp =
          Bmat + bOff + (size_t)(n0 + bn) * ldb + k0 + bq * 8;
      unsigned lo = (unsigned)(uintptr_t)&Bs[buf][bn * 32 + bq * 8];
      async_g2l_b128(lo, gp);
    }
  };

  constexpr int NPS = A_F32 ? 1 : 3;   // async ops per thread per stage
  const int nsteps = K >> 5;

  issue(0, 0);
  for (int i = 0; i < nsteps; ++i) {
    const int buf = i & 1;
    if (i + 1 < nsteps) {
      issue((i + 1) << 5, buf ^ 1);
      wait_asynccnt<NPS>();            // stage i's async copies complete
    } else {
      wait_asynccnt<0>();
    }
    __syncthreads();

    // fragments per documented 16-bit WMMA layout:
    // lane<16: K 0..7 in V0-3, 16..23 in V4-7 ; lane>=16: +8
    union Frag { v16bf v; unsigned u[8]; };
    const int kb = (lane >> 4) << 3;
    const int rr = lane & 15;
    Frag a[2], b[2];
#pragma unroll
    for (int mi = 0; mi < 2; ++mi) {
      const int r = waveM * 32 + mi * 16 + rr;
#pragma unroll
      for (int j = 0; j < 4; ++j) {
        a[mi].u[j]     = *(const unsigned*)&As[buf][r * 32 + kb + 2 * j];
        a[mi].u[j + 4] = *(const unsigned*)&As[buf][r * 32 + 16 + kb + 2 * j];
      }
    }
#pragma unroll
    for (int ni = 0; ni < 2; ++ni) {
      const int c = waveN * 32 + ni * 16 + rr;
#pragma unroll
      for (int j = 0; j < 4; ++j) {
        b[ni].u[j]     = *(const unsigned*)&Bs[buf][c * 32 + kb + 2 * j];
        b[ni].u[j + 4] = *(const unsigned*)&Bs[buf][c * 32 + 16 + kb + 2 * j];
      }
    }
#pragma unroll
    for (int mi = 0; mi < 2; ++mi)
#pragma unroll
      for (int ni = 0; ni < 2; ++ni)
        acc[mi][ni] = __builtin_amdgcn_wmma_f32_16x16x32_bf16(
            false, a[mi].v, false, b[ni].v, (short)0, acc[mi][ni], false, false);
    __syncthreads();   // LDS buffer free for reuse
  }

  // ---- epilogue ----
  const int rbase = 8 * (lane >> 4);
  const int cloc  = lane & 15;
  (void)mask; (void)bias;
#pragma unroll
  for (int mi = 0; mi < 2; ++mi)
#pragma unroll
    for (int ni = 0; ni < 2; ++ni)
#pragma unroll
      for (int r = 0; r < 8; ++r) {
        const int row = m0 + waveM * 32 + mi * 16 + rbase + r;
        const int col = n0 + waveN * 32 + ni * 16 + cloc;
        const float va = acc[mi][ni][r];
        if (MODE == MODE_QK) {
          const float val = va + bias[col];
          const int bi = row >> 11, s = row & (S_ - 1);
          const int h = col >> 6, dh = col & 63;
          ((unsigned short*)outv)[((((size_t)bi * H_ + h) * S_ + s) << 6) + dh] =
              f2bf(val);
        } else if (MODE == MODE_VT) {
          const float val = va + bias[col];
          const int bi = row >> 11, s = row & (S_ - 1);
          const int h = col >> 6, dh = col & 63;
          // transposed head-split: [B,H,DH,S]
          ((unsigned short*)outv)[(((size_t)bi * H_ + h) * DH_ + dh) * S_ + s] =
              f2bf(val);
        } else if (MODE == MODE_OUTPROJ) {
          ((float*)outv)[(size_t)row * D_ + col] = va + bias[col];
        } else if (MODE == MODE_SCORES) {
          const int bz = z >> 4;
          const size_t midx = ((size_t)bz * S_ + row) * S_ + col;
          const float val = mask[midx] ? va * 0.125f : -1e9f;  // 1/sqrt(64)
          ((float*)outv)[((size_t)z * S_ + row) * S_ + col] = val;
        } else {  // MODE_CONTEXT
          const int bz = z >> 4, h = z & 15;
          ((unsigned short*)outv)[((size_t)(bz * S_ + row)) * D_ + h * 64 + col] =
              f2bf(va);
        }
      }
}

// ---------------------------------------------------------------------------
// Host-side orchestration
// ---------------------------------------------------------------------------
extern "C" void kernel_launch(void* const* d_in, const int* in_sizes, int n_in,
                              void* d_out, int out_size, void* d_ws, size_t ws_size,
                              hipStream_t stream) {
  (void)in_sizes; (void)n_in; (void)out_size; (void)ws_size;

  const float* q_in = (const float*)d_in[0];
  const float* k_in = (const float*)d_in[1];
  const float* v_in = (const float*)d_in[2];
  const unsigned char* mask = (const unsigned char*)d_in[3];
  const float* Wq = (const float*)d_in[4];  const float* bq = (const float*)d_in[5];
  const float* Wk = (const float*)d_in[6];  const float* bk = (const float*)d_in[7];
  const float* Wv = (const float*)d_in[8];  const float* bv = (const float*)d_in[9];
  const float* Wo = (const float*)d_in[10]; const float* bo = (const float*)d_in[11];

  float* outp  = (float*)d_out;                              // [B,S,D]
  float* attnp = outp + (size_t)B_ * S_ * D_;                // [B,H,S,S]

  // workspace layout
  char* ws = (char*)d_ws;
  const size_t XBYTES = (size_t)B_ * S_ * D_ * 2;   // 8 MiB per bf16 activation
  const size_t WBYTES = (size_t)D_ * D_ * 2;        // 2 MiB per bf16 weight
  unsigned short* Xq  = (unsigned short*)(ws + 0 * XBYTES);
  unsigned short* Xk  = (unsigned short*)(ws + 1 * XBYTES);
  unsigned short* Xv  = (unsigned short*)(ws + 2 * XBYTES);
  unsigned short* Wqt = (unsigned short*)(ws + 3 * XBYTES + 0 * WBYTES);
  unsigned short* Wkt = (unsigned short*)(ws + 3 * XBYTES + 1 * WBYTES);
  unsigned short* Wvt = (unsigned short*)(ws + 3 * XBYTES + 2 * WBYTES);
  unsigned short* Wot = (unsigned short*)(ws + 3 * XBYTES + 3 * WBYTES);
  unsigned short* Qh  = (unsigned short*)(ws + 3 * XBYTES + 4 * WBYTES + 0 * XBYTES);
  unsigned short* Kh  = (unsigned short*)(ws + 3 * XBYTES + 4 * WBYTES + 1 * XBYTES);
  unsigned short* Vht = (unsigned short*)(ws + 3 * XBYTES + 4 * WBYTES + 2 * XBYTES);
  unsigned short* Ctx = (unsigned short*)(ws + 6 * XBYTES + 4 * WBYTES);

  // 1) casts: activations row-major, weights transposed to [N][K]
  {
    const int nx4 = (B_ * S_ * D_) / 4;   // 1,048,576
    cast_f32_bf16<<<nx4 / 256, 256, 0, stream>>>(q_in, Xq, nx4);
    cast_f32_bf16<<<nx4 / 256, 256, 0, stream>>>(k_in, Xk, nx4);
    cast_f32_bf16<<<nx4 / 256, 256, 0, stream>>>(v_in, Xv, nx4);
    dim3 tg(D_ / 32, D_ / 32, 1);
    cast_transpose_w<<<tg, 256, 0, stream>>>(Wq, Wqt);
    cast_transpose_w<<<tg, 256, 0, stream>>>(Wk, Wkt);
    cast_transpose_w<<<tg, 256, 0, stream>>>(Wv, Wvt);
    cast_transpose_w<<<tg, 256, 0, stream>>>(Wo, Wot);
  }

  // 2) QKV projections
  {
    dim3 grid(D_ / 64, (B_ * S_) / 128, 1);
    gemm_wmma<MODE_QK, false><<<grid, 256, 0, stream>>>(
        Xq, Wqt, bq, nullptr, Qh, B_ * S_, D_, D_, D_, D_);
    gemm_wmma<MODE_QK, false><<<grid, 256, 0, stream>>>(
        Xk, Wkt, bk, nullptr, Kh, B_ * S_, D_, D_, D_, D_);
    gemm_wmma<MODE_VT, false><<<grid, 256, 0, stream>>>(
        Xv, Wvt, bv, nullptr, Vht, B_ * S_, D_, D_, D_, D_);
  }

  // 3) scores = scale * Q K^T (+ mask) -> attn (f32); B = Kh is [S][DH] n-major
  {
    dim3 grid(S_ / 64, S_ / 128, B_ * H_);
    gemm_wmma<MODE_SCORES, false><<<grid, 256, 0, stream>>>(
        Qh, Kh, nullptr, mask, attnp, S_, S_, DH_, DH_, DH_);
  }

  // 4) softmax in place over each of B*H*S rows
  softmax_rows<<<B_ * H_ * S_, 256, 0, stream>>>(attnp);

  // 5) context = attn @ V -> bf16 [B,S,D]; B = Vht is [DH][S] n-major
  {
    dim3 grid(DH_ / 64, S_ / 128, B_ * H_);
    gemm_wmma<MODE_CONTEXT, true><<<grid, 256, 0, stream>>>(
        attnp, Vht, nullptr, nullptr, Ctx, S_, DH_, S_, S_, S_);
  }

  // 6) output = context @ Wo + bo -> f32 [B,S,D]
  {
    dim3 grid(D_ / 64, (B_ * S_) / 128, 1);
    gemm_wmma<MODE_OUTPROJ, false><<<grid, 256, 0, stream>>>(
        Ctx, Wot, bo, nullptr, outp, B_ * S_, D_, D_, D_, D_);
  }
}
